// Encoder_51582557225692
// MI455X (gfx1250) — matrix operand order repaired and checked by reading, not compiled, not run on hardware
//
#include <hip/hip_runtime.h>
#include <hip/hip_bf16.h>
#include <math.h>

// Problem constants (from reference)
#define LCONST 2
#define BCONST 4
#define HCONST 4096
#define NHCONST 32
#define HDCONST 128
#define SCONST 64
#define ICONST 11008

typedef __attribute__((ext_vector_type(16))) __bf16 v16bf;
typedef __attribute__((ext_vector_type(8)))  float  v8f;

// ---------------------------------------------------------------------------
// Software-pipelined tiled GEMM: C[M,N] (+)= A[M,K] * W[K,N]
// fp32 in global, bf16 in (double-buffered) LDS, v_wmma_f32_16x16x32_bf16.
// lda==K, ldw==N. Row remap on store:
//   b = srcRow / rpb_src; dstRow = b*rpb_dst + dst_off + srcRow%rpb_src
// ---------------------------------------------------------------------------
#define BM 128
#define BN 128
#define BK 32
#define BKP 36   // padded bf16 row stride (avoids LDS bank conflicts)

__global__ __launch_bounds__(256) void gemm_bf16_wmma(
    const float* __restrict__ A, const float* __restrict__ W, float* __restrict__ C,
    int N, int K, int rpb_src, int rpb_dst, int dst_off, int ldc, int accumulate)
{
    __shared__ __bf16 As[2][BM * BKP];
    __shared__ __bf16 Bs[2][BN * BKP];

    const int tid  = threadIdx.x;
    const int wave = tid >> 5;
    const int lane = tid & 31;
    const int tileN = blockIdx.x * BN;
    const int tileM = blockIdx.y * BM;

    v8f acc[8];
#pragma unroll
    for (int j = 0; j < 8; ++j) acc[j] = (v8f){0,0,0,0,0,0,0,0};

    const int mrow = wave * 16;          // this wave's 16-row strip in the tile
    const int lm   = lane & 15;
    const int kgA  = (lane >> 4) * 16;   // A frag K-group (contiguous k = 16h + s, matches B)
    const int kgB  = (lane >> 4) * 16;   // B frag K-group (contiguous k = 16h + s, matches A)

    // Per-thread staging assignments (constant across iterations)
    // A tile: 128x32 fp32 = 1024 float4; thread handles 4 (f = tid + i*256)
    // W tile: 32x128 fp32 = 1024 float4; thread handles 4
    int amr[4], ac4[4], wkk[4], wn4[4];
#pragma unroll
    for (int i = 0; i < 4; ++i) {
        int f  = tid + i * 256;
        amr[i] = f >> 3;            // A row (8 float4 per 32-col row)
        ac4[i] = (f & 7) * 4;       // A col
        wkk[i] = f >> 5;            // W row (32 float4 per 128-col row)
        wn4[i] = (f & 31) * 4;      // W col
    }

    float4 va[4], vb[4];

    // issue ALL global loads for tile k0 before any wait
    auto load_tile = [&](int k0) {
#pragma unroll
        for (int i = 0; i < 4; ++i)
            va[i] = *(const float4*)(A + (size_t)(tileM + amr[i]) * K + k0 + ac4[i]);
#pragma unroll
        for (int i = 0; i < 4; ++i)
            vb[i] = *(const float4*)(W + (size_t)(k0 + wkk[i]) * N + tileN + wn4[i]);
    };
    // convert + store staged tile into LDS buffer `buf`
    auto store_tile = [&](int buf) {
#pragma unroll
        for (int i = 0; i < 4; ++i) {
            __bf16* d = &As[buf][amr[i] * BKP + ac4[i]];
            d[0] = (__bf16)va[i].x; d[1] = (__bf16)va[i].y;
            d[2] = (__bf16)va[i].z; d[3] = (__bf16)va[i].w;
        }
#pragma unroll
        for (int i = 0; i < 4; ++i) {
            Bs[buf][(wn4[i] + 0) * BKP + wkk[i]] = (__bf16)vb[i].x;
            Bs[buf][(wn4[i] + 1) * BKP + wkk[i]] = (__bf16)vb[i].y;
            Bs[buf][(wn4[i] + 2) * BKP + wkk[i]] = (__bf16)vb[i].z;
            Bs[buf][(wn4[i] + 3) * BKP + wkk[i]] = (__bf16)vb[i].w;
        }
    };

    const int nTiles = K / BK;

    // prologue: stage tile 0
    load_tile(0);
    store_tile(0);
    __syncthreads();

    for (int t = 0; t < nTiles; ++t) {
        const int cur = t & 1;
        const bool more = (t + 1) < nTiles;

        // issue next tile's global loads (no wait yet)
        if (more) load_tile((t + 1) * BK);
        else if (t + 1 < nTiles) ; // keep compiler honest
        if (more)
            __builtin_prefetch(W + (size_t)((t + 2) * BK % K + wkk[0]) * N + tileN + wn4[0], 0, 0);

        // fragment loads from current LDS buffer (hoisted so WMMAs issue densely)
        v16bf af;
        {
            const __bf16* p = &As[cur][(mrow + lm) * BKP];
#pragma unroll
            for (int e = 0; e < 8; ++e) af[e]     = p[kgA + e];
#pragma unroll
            for (int e = 0; e < 8; ++e) af[8 + e] = p[kgA + 8 + e];
        }
        v16bf bfm[8];
#pragma unroll
        for (int j = 0; j < 8; ++j) {
            const __bf16* p = &Bs[cur][(j * 16 + lm) * BKP + kgB];
#pragma unroll
            for (int e = 0; e < 16; ++e) bfm[j][e] = p[e];
        }
#pragma unroll
        for (int j = 0; j < 8; ++j)
            acc[j] = __builtin_amdgcn_wmma_f32_16x16x32_bf16(
                false, af, false, bfm[j], (short)0, acc[j], false, false);

        // drain next tile into the other LDS buffer while WMMAs retire
        if (more) store_tile(cur ^ 1);
        __syncthreads();
    }

    // ---- epilogue: C layout (lanes 0-15: N=lane, M=r; lanes 16-31: N=lane-16, M=8+r) ----
    const int rbase = tileM + mrow + (lane >> 4) * 8;
#pragma unroll
    for (int j = 0; j < 8; ++j) {
        int col = tileN + j * 16 + lm;
#pragma unroll
        for (int r = 0; r < 8; ++r) {
            int srow = rbase + r;
            int b    = srow / rpb_src;
            int drow = b * rpb_dst + dst_off + (srow % rpb_src);
            float* cp = C + (size_t)drow * ldc + col;
            if (accumulate) *cp = *cp + acc[j][r];
            else            *cp = acc[j][r];
        }
    }
}

// ---------------------------------------------------------------------------
// RMSNorm: y[row] = x[row] * rsqrt(mean(x^2)+eps) * w     (one block per row)
// ---------------------------------------------------------------------------
__global__ __launch_bounds__(256) void rmsnorm_k(
    const float* __restrict__ x, const float* __restrict__ w, float* __restrict__ y, int cols)
{
    int row = blockIdx.x;
    const float* xr = x + (size_t)row * cols;
    float ss = 0.f;
    for (int c = threadIdx.x; c < cols; c += 256) { float v = xr[c]; ss += v * v; }
    __shared__ float red[256];
    red[threadIdx.x] = ss; __syncthreads();
    for (int s = 128; s > 0; s >>= 1) {
        if (threadIdx.x < s) red[threadIdx.x] += red[threadIdx.x + s];
        __syncthreads();
    }
    float inv = rsqrtf(red[0] / (float)cols + 1e-5f);
    float* yr = y + (size_t)row * cols;
    for (int c = threadIdx.x; c < cols; c += 256) yr[c] = xr[c] * inv * w[c];
}

// ---------------------------------------------------------------------------
// RoPE in place: row r, pos = pos_off + r % seq_per_b; per-head half rotation
// ---------------------------------------------------------------------------
__global__ __launch_bounds__(256) void rope_k(float* __restrict__ x, int seq_per_b, int pos_off)
{
    int row = blockIdx.x;
    int pos = pos_off + (row % seq_per_b);
    float* xr = x + (size_t)row * HCONST;
    const int half = HDCONST / 2;
    for (int idx = threadIdx.x; idx < HCONST / 2; idx += 256) {
        int head = idx / half;
        int i    = idx % half;
        float inv = __expf(-9.210340371976184f * (float)i / (float)half); // 10000^{-2i/HD}
        float ang = (float)pos * inv;
        float c = __cosf(ang), s = __sinf(ang);
        int d0 = head * HDCONST + i;
        int d1 = d0 + half;
        float x0 = xr[d0], x1 = xr[d1];
        xr[d0] = x0 * c - x1 * s;
        xr[d1] = x1 * c + x0 * s;
    }
}

// ---------------------------------------------------------------------------
// Attention: one block per (b, h, qi). Block = 192 threads (one per key).
// ---------------------------------------------------------------------------
__global__ __launch_bounds__(192) void attention_k(
    const float* __restrict__ q, const float* __restrict__ kc,
    const float* __restrict__ vc, float* __restrict__ o)
{
    const int Q = 2 * SCONST, KEYS = 3 * SCONST;
    int gid = blockIdx.x;
    int qi  = gid % Q;
    int h   = (gid / Q) % NHCONST;
    int b   = gid / (Q * NHCONST);
    int t   = threadIdx.x;

    __shared__ float qs[HDCONST];
    __shared__ float p[3 * SCONST];
    __shared__ float mxsm[2];

    const float* qrow = q + ((size_t)(b * Q + qi) * HCONST) + h * HDCONST;
    if (t < HDCONST) qs[t] = qrow[t];
    __syncthreads();

    // score for key t
    const float* krow = kc + ((size_t)(b * KEYS + t) * HCONST) + h * HDCONST;
    float dot = 0.f;
    for (int d = 0; d < HDCONST; ++d) dot += qs[d] * krow[d];
    float sc = dot * 0.08838834764831845f; // 1/sqrt(128)
    {
        bool memk = t < SCONST;
        bool ordk = (t >= SCONST) && (t < 2 * SCONST);
        bool beak = t >= 2 * SCONST;
        bool allow = (qi < SCONST)
            ? (memk || (ordk && (t - SCONST <= qi)))
            : (memk || ordk || (beak && (t - 2 * SCONST <= qi - SCONST)));
        if (!allow) sc = -3.0e38f;
    }
    p[t] = sc; __syncthreads();
    if (t == 0) { float m = -3.4e38f; for (int j = 0; j < KEYS; ++j) m = fmaxf(m, p[j]); mxsm[0] = m; }
    __syncthreads();
    float e = __expf(sc - mxsm[0]);
    p[t] = e; __syncthreads();
    if (t == 0) { float s = 0.f; for (int j = 0; j < KEYS; ++j) s += p[j]; mxsm[1] = s; }
    __syncthreads();
    float invs = 1.f / mxsm[1];
    if (t < HDCONST) {
        float a = 0.f;
        for (int j = 0; j < KEYS; ++j)
            a += p[j] * vc[((size_t)(b * KEYS + j) * HCONST) + h * HDCONST + t];
        o[((size_t)(b * Q + qi) * HCONST) + h * HDCONST + t] = a * invs;
    }
}

// ---------------------------------------------------------------------------
// g = silu(g) * u  (in place on g)
// ---------------------------------------------------------------------------
__global__ __launch_bounds__(256) void silu_gate_k(float* __restrict__ g, const float* __restrict__ u, size_t n)
{
    size_t i = (size_t)blockIdx.x * 256 + threadIdx.x;
    if (i < n) {
        float x = g[i];
        g[i] = (x / (1.f + __expf(-x))) * u[i];
    }
}

// ---------------------------------------------------------------------------
// x rows: [embed gather | beacon]  (one block per row of (B, 2S, H))
// ---------------------------------------------------------------------------
__global__ __launch_bounds__(256) void assemble_x_k(
    const int* __restrict__ ids, const float* __restrict__ embed,
    const float* __restrict__ beacon, float* __restrict__ x)
{
    int row = blockIdx.x;
    int b = row / (2 * SCONST);
    int s = row % (2 * SCONST);
    const float* src;
    if (s < SCONST) src = embed + (size_t)ids[b * SCONST + s] * HCONST;
    else            src = beacon + ((size_t)b * SCONST + (s - SCONST)) * HCONST;
    float* dst = x + (size_t)row * HCONST;
    for (int c = threadIdx.x; c < HCONST; c += 256) dst[c] = src[c];
}

__global__ __launch_bounds__(256) void copy_flat_k(const float* __restrict__ src, float* __restrict__ dst, size_t n)
{
    size_t i = (size_t)blockIdx.x * 256 + threadIdx.x;
    if (i < n) dst[i] = src[i];
}

// out[B*S*H ..] = x[:, S:2S]  (one block per (b,s) row)
__global__ __launch_bounds__(256) void copy_bs_out_k(const float* __restrict__ x, float* __restrict__ out)
{
    int row = blockIdx.x;            // 0 .. B*S-1
    int b = row / SCONST;
    int s = row % SCONST;
    const float* src = x + ((size_t)(b * 2 * SCONST + SCONST + s)) * HCONST;
    float* dst = out + ((size_t)BCONST * SCONST * HCONST) + (size_t)row * HCONST;
    for (int c = threadIdx.x; c < HCONST; c += 256) dst[c] = src[c];
}

// ---------------------------------------------------------------------------
// Host orchestration. Only layer 0 feeds the output (updates are appended
// BEFORE each layer's compute), so layer 1 is dead code and skipped.
// ---------------------------------------------------------------------------
extern "C" void kernel_launch(void* const* d_in, const int* in_sizes, int n_in,
                              void* d_out, int out_size, void* d_ws, size_t ws_size,
                              hipStream_t stream)
{
    (void)in_sizes; (void)n_in; (void)out_size; (void)ws_size;
    const int*   ids    = (const int*)  d_in[0];
    const float* memory = (const float*)d_in[1];   // (L*B, S, H); layer 0 = first B*S rows
    const float* beacon = (const float*)d_in[2];
    const float* embed  = (const float*)d_in[3];
    const float* Wq  = (const float*)d_in[4];      // layer 0 slices = base pointers
    const float* Wk  = (const float*)d_in[5];
    const float* Wv  = (const float*)d_in[6];
    const float* Wo  = (const float*)d_in[7];
    const float* Wkm = (const float*)d_in[8];
    const float* Wvm = (const float*)d_in[9];
    const float* ln1 = (const float*)d_in[10];
    const float* ln2 = (const float*)d_in[11];
    const float* Wg  = (const float*)d_in[12];
    const float* Wu  = (const float*)d_in[13];
    const float* Wd  = (const float*)d_in[14];
    float* out = (float*)d_out;
    float* ws  = (float*)d_ws;

    const int H = HCONST, I = ICONST, Sq = 2 * SCONST;            // 128
    const int M  = BCONST * Sq;                                    // 512
    const int Mm = BCONST * SCONST;                                // 256
    const int Mk = BCONST * 3 * SCONST;                            // 768

    // workspace layout (floats)
    float* x    = ws;                                    // 512*4096
    float* h    = x    + (size_t)M  * H;                 // 512*4096 (h1 / attn out / h2)
    float* q    = h    + (size_t)M  * H;                 // 512*4096
    float* kcat = q    + (size_t)M  * H;                 // 768*4096
    float* vcat = kcat + (size_t)Mk * H;                 // 768*4096
    float* g    = vcat + (size_t)Mk * H;                 // 512*11008
    float* u    = g    + (size_t)M  * I;                 // 512*11008

    // 1) x = [embed[ids] | beacon]; out[0:BSH] = beacon
    assemble_x_k<<<M, 256, 0, stream>>>(ids, embed, beacon, x);
    {
        size_t n = (size_t)BCONST * SCONST * H;
        copy_flat_k<<<(unsigned)((n + 255) / 256), 256, 0, stream>>>(beacon, out, n);
    }

    // 2) h1 = rms(x, ln1[0])
    rmsnorm_k<<<M, 256, 0, stream>>>(x, ln1, h, H);

    // 3) q/k/v projections (k,v written straight into concat buffers)
    dim3 gM(H / BN, M / BM), gMm(H / BN, Mm / BM);
    gemm_bf16_wmma<<<gM, 256, 0, stream>>>(h, Wq, q,    H, H, M,   M,           0,      H, 0);
    gemm_bf16_wmma<<<gM, 256, 0, stream>>>(h, Wk, kcat, H, H, Sq,  3 * SCONST,  SCONST, H, 0);
    gemm_bf16_wmma<<<gM, 256, 0, stream>>>(h, Wv, vcat, H, H, Sq,  3 * SCONST,  SCONST, H, 0);
    gemm_bf16_wmma<<<gMm,256, 0, stream>>>(memory, Wkm, kcat, H, H, SCONST, 3 * SCONST, 0, H, 0);
    gemm_bf16_wmma<<<gMm,256, 0, stream>>>(memory, Wvm, vcat, H, H, SCONST, 3 * SCONST, 0, H, 0);

    // 4) RoPE: q positions S..3S-1, k positions 0..3S-1
    rope_k<<<M,  256, 0, stream>>>(q,    Sq,         SCONST);
    rope_k<<<Mk, 256, 0, stream>>>(kcat, 3 * SCONST, 0);

    // 5) attention -> h (reused as o)
    attention_k<<<BCONST * NHCONST * Sq, 192, 0, stream>>>(q, kcat, vcat, h);

    // 6) x += o @ Wo
    gemm_bf16_wmma<<<gM, 256, 0, stream>>>(h, Wo, x, H, H, M, M, 0, H, 1);

    // 7) h2 = rms(x, ln2[0])
    rmsnorm_k<<<M, 256, 0, stream>>>(x, ln2, h, H);

    // 8) MLP: g = h2@Wg, u = h2@Wu, g = silu(g)*u, x += g@Wd
    dim3 gI(I / BN, M / BM), gD(H / BN, M / BM);
    gemm_bf16_wmma<<<gI, 256, 0, stream>>>(h, Wg, g, I, H, M, M, 0, I, 0);
    gemm_bf16_wmma<<<gI, 256, 0, stream>>>(h, Wu, u, I, H, M, M, 0, I, 0);
    {
        size_t n = (size_t)M * I;
        silu_gate_k<<<(unsigned)((n + 255) / 256), 256, 0, stream>>>(g, u, n);
    }
    gemm_bf16_wmma<<<gD, 256, 0, stream>>>(g, Wd, x, H, I, M, M, 0, H, 1);

    // 9) out[BSH:2BSH] = x[:, S:2S]  (bs after layer 0)
    copy_bs_out_k<<<BCONST * SCONST, 256, 0, stream>>>(x, out);
}
